// HierarchicalMetaAttention_25220047962725
// MI455X (gfx1250) — matrix operand order repaired and checked by reading, not compiled, not run on hardware
//
#include <hip/hip_runtime.h>
#include <math.h>
#include <stdint.h>

// ---------------------------------------------------------------------------
// CDNA5 (gfx1250) types & helpers
// ---------------------------------------------------------------------------
typedef __bf16 bf16;
typedef bf16  v16bf __attribute__((ext_vector_type(16)));
typedef bf16  v8bf  __attribute__((ext_vector_type(8)));
typedef float v8f   __attribute__((ext_vector_type(8)));

__device__ __forceinline__ bf16 f2bf(float f) { return (bf16)f; }

__device__ __forceinline__ v8f wmma_bf16(v16bf a, v16bf b, v8f c) {
  // D = A(16x32 bf16) * B(32x16 bf16) + C(16x16 f32)
  return __builtin_amdgcn_wmma_f32_16x16x32_bf16(false, a, false, b, (short)0, c,
                                                 false, false);
}

__device__ __forceinline__ v8f v8f_zero() {
  v8f z;
#pragma unroll
  for (int i = 0; i < 8; ++i) z[i] = 0.0f;
  return z;
}

// 16-byte global -> LDS copy. Async (ASYNCcnt-tracked, bypasses VGPRs) on
// gfx1250; generic LDS pointer's low 32 bits are the LDS byte address.
#define USE_ASYNC_LDS 1

__device__ __forceinline__ void cp_lds16(bf16* dst, const bf16* src) {
#if USE_ASYNC_LDS
  unsigned l = (unsigned)(unsigned long long)(uintptr_t)dst;
  unsigned long long g = (unsigned long long)(uintptr_t)src;
  asm volatile("global_load_async_to_lds_b128 %0, %1, off"
               :: "v"(l), "v"(g) : "memory");
#else
  *(v8bf*)dst = *(const v8bf*)src;
#endif
}
__device__ __forceinline__ void async_wait() {
#if USE_ASYNC_LDS
  asm volatile("s_wait_asynccnt 0" ::: "memory");
#endif
}

// ---------------------------------------------------------------------------
// Generic bf16-WMMA GEMM:  C[M,N] = act( A[M,K] @ W[N,K]^T + bias + Cin )
// block = 256 threads (8 waves), tile 128x64, K-step 32. Optional bf16 output.
// ---------------------------------------------------------------------------
#define GM 128
#define GN 64
#define GK 32

__global__ __launch_bounds__(256) void gemm_bias_act_kernel(
    const float* __restrict__ A, const float* __restrict__ W,
    const float* __restrict__ bias, const float* __restrict__ Cin,
    float* __restrict__ C, int M, int N, int K, int lda, int ldw, int ldc,
    long strideA, long strideC, int act, int outBf16) {
  __shared__ __align__(16) bf16 As[GM][GK + 8];
  __shared__ __align__(16) bf16 Bs[GN][GK + 8];

  const int tid  = threadIdx.x;
  const int wid  = tid >> 5;
  const int lane = tid & 31;
  const int wrow = wid >> 1;     // 0..3
  const int wcol = wid & 1;      // 0..1
  const int m16  = lane & 15;
  const int hi   = lane >> 4;    // 0/1
  const int z    = blockIdx.z;

  const float* Ab   = A + (long)z * strideA;
  float*       Cb   = C + (long)z * strideC;
  bf16*        Cb16 = (bf16*)C + (long)z * strideC;
  const float* Cinb = Cin ? (Cin + (long)z * strideC) : nullptr;

  const int row0 = blockIdx.y * GM;
  const int col0 = blockIdx.x * GN;

  v8f cf[2][2];
  cf[0][0] = v8f_zero(); cf[0][1] = v8f_zero();
  cf[1][0] = v8f_zero(); cf[1][1] = v8f_zero();

  for (int k0 = 0; k0 < K; k0 += GK) {
    // ---- stage A tile (128x32 f32 -> bf16), coalesced float4 ----
#pragma unroll
    for (int it = 0; it < 4; ++it) {
      int q  = tid + 256 * it;          // 0..1023
      int r  = q >> 3;                  // row in tile
      int kq = (q & 7) * 4;             // k offset
      float4 v = make_float4(0.f, 0.f, 0.f, 0.f);
      if (row0 + r < M)
        v = *(const float4*)(Ab + (long)(row0 + r) * lda + k0 + kq);
      As[r][kq + 0] = f2bf(v.x); As[r][kq + 1] = f2bf(v.y);
      As[r][kq + 2] = f2bf(v.z); As[r][kq + 3] = f2bf(v.w);
    }
    // ---- stage B tile (64x32 from W[N,K]) ----
#pragma unroll
    for (int it = 0; it < 2; ++it) {
      int q  = tid + 256 * it;          // 0..511
      int n  = q >> 3;
      int kq = (q & 7) * 4;
      float4 v = make_float4(0.f, 0.f, 0.f, 0.f);
      if (col0 + n < N)
        v = *(const float4*)(W + (long)(col0 + n) * ldw + k0 + kq);
      Bs[n][kq + 0] = f2bf(v.x); Bs[n][kq + 1] = f2bf(v.y);
      Bs[n][kq + 2] = f2bf(v.z); Bs[n][kq + 3] = f2bf(v.w);
    }
    // ---- prefetch next K-tile into cache (global_prefetch_b8) ----
    if (k0 + GK < K) {
      if (tid < 128) {
        int pr = tid;
        if (row0 + pr < M)
          __builtin_prefetch(Ab + (long)(row0 + pr) * lda + k0 + GK, 0, 0);
      } else {
        int pn = tid - 128;
        if (pn < GN && col0 + pn < N)
          __builtin_prefetch(W + (long)(col0 + pn) * ldw + k0 + GK, 0, 0);
      }
    }
    __syncthreads();

    // ---- fragments (contiguous k = 16*hi + e for A and B) ----
    const int kb = hi * 8;
    const int kd = hi * 16;
    v16bf a0, a1, b0, b1;
#pragma unroll
    for (int e = 0; e < 8; ++e) {
      a0[e]     = As[wrow * 32 + m16][kd + e];
      a0[e + 8] = As[wrow * 32 + m16][kd + 8 + e];
      a1[e]     = As[wrow * 32 + 16 + m16][kd + e];
      a1[e + 8] = As[wrow * 32 + 16 + m16][kd + 8 + e];
    }
#pragma unroll
    for (int e = 0; e < 16; ++e) {
      b0[e] = Bs[wcol * 32 + m16][kd + e];
      b1[e] = Bs[wcol * 32 + 16 + m16][kd + e];
    }
    cf[0][0] = wmma_bf16(a0, b0, cf[0][0]);
    cf[0][1] = wmma_bf16(a0, b1, cf[0][1]);
    cf[1][0] = wmma_bf16(a1, b0, cf[1][0]);
    cf[1][1] = wmma_bf16(a1, b1, cf[1][1]);
    __syncthreads();
  }

  // ---- epilogue ----
  const int mbase = hi * 8;
#pragma unroll
  for (int ti = 0; ti < 2; ++ti) {
#pragma unroll
    for (int tj = 0; tj < 2; ++tj) {
      int gcol = col0 + wcol * 32 + tj * 16 + m16;
      if (gcol < N) {
        float bv = bias ? bias[gcol] : 0.0f;
#pragma unroll
        for (int r = 0; r < 8; ++r) {
          int grow = row0 + wrow * 32 + ti * 16 + mbase + r;
          if (grow < M) {
            float vv = cf[ti][tj][r] + bv;
            if (Cinb) vv += Cinb[(long)grow * ldc + gcol];
            if (act == 1) vv = fmaxf(vv, 0.0f);
            else if (act == 2) vv = 1.0f / (1.0f + __expf(-vv));
            if (outBf16) Cb16[(long)grow * ldc + gcol] = f2bf(vv);
            else         Cb[(long)grow * ldc + gcol]   = vv;
          }
        }
      }
    }
  }
}

// ---------------------------------------------------------------------------
// Flash attention (bf16 WMMA, online softmax). Q/K/V are bf16, O is f32.
// grid = (S/64, nHeads, batch), block = 128 (4 waves, 16 q-rows each), hd = 64
// K/V tiles (32x64) staged block-wide into LDS via async global->LDS copies.
// ---------------------------------------------------------------------------
__global__ __launch_bounds__(128) void flash_attn_kernel(
    const bf16* __restrict__ Q, const bf16* __restrict__ K,
    const bf16* __restrict__ V, float* __restrict__ O, int ldq, int ldk,
    int ldv, int ldo, long sq, long sk, long sv, long so, int S, float scale) {
  __shared__ __align__(16) bf16 Kt[32][72];
  __shared__ __align__(16) bf16 Vt[32][72];
  __shared__ float sS[4][16][36];
  __shared__ __align__(16) bf16 sP[4][16][40];
  __shared__ float sAlpha[4][16];
  __shared__ float sL[4][16];

  const int tid  = threadIdx.x;
  const int wid  = tid >> 5;
  const int lane = tid & 31;
  const int h    = blockIdx.y;
  const int z    = blockIdx.z;
  const int r0   = blockIdx.x * 64 + wid * 16;
  const int m16  = lane & 15;
  const int hi   = lane >> 4;
  const int mbase = hi * 8;

  const bf16* q = Q + (long)z * sq + (long)h * 64;
  const bf16* k = K + (long)z * sk + (long)h * 64;
  const bf16* v = V + (long)z * sv + (long)h * 64;
  float*      o = O + (long)z * so + (long)h * 64;

  // Q fragments (16 rows x 64 d, two K=32 halves) — 16B vector loads
  v16bf aq0, aq1;
  {
    const bf16* qr = q + (long)(r0 + m16) * ldq;
    const int kb = hi * 16;
    v8bf x0 = *(const v8bf*)(qr + kb);
    v8bf x1 = *(const v8bf*)(qr + kb + 8);
    v8bf x2 = *(const v8bf*)(qr + 32 + kb);
    v8bf x3 = *(const v8bf*)(qr + 32 + kb + 8);
#pragma unroll
    for (int e = 0; e < 8; ++e) {
      aq0[e] = x0[e]; aq0[e + 8] = x1[e];
      aq1[e] = x2[e]; aq1[e + 8] = x3[e];
    }
  }

  float mrow = -3.0e38f, lrow = 0.0f;   // valid in lanes 0..15 (row = lane)
  v8f acc[4];
#pragma unroll
  for (int cg = 0; cg < 4; ++cg) acc[cg] = v8f_zero();

  for (int kb0 = 0; kb0 < S; kb0 += 32) {
    // ---- stage K & V tiles (32 rows x 64 bf16 each) block-wide ----
#pragma unroll
    for (int it = 0; it < 2; ++it) {
      int c   = tid + 128 * it;       // 0..255
      int row = c >> 3;               // 0..31
      int seg = c & 7;                // 16B segment in row
      cp_lds16(&Kt[row][seg * 8], k + (long)(kb0 + row) * ldk + seg * 8);
      cp_lds16(&Vt[row][seg * 8], v + (long)(kb0 + row) * ldv + seg * 8);
    }
    async_wait();
    __syncthreads();

    // ---- scores 16x32 via WMMA ----
    const int kd = hi * 16;
#pragma unroll
    for (int jn = 0; jn < 2; ++jn) {
      int key = jn * 16 + m16;
      v8bf t0 = *(const v8bf*)&Kt[key][kd];
      v8bf t1 = *(const v8bf*)&Kt[key][kd + 8];
      v8bf t2 = *(const v8bf*)&Kt[key][32 + kd];
      v8bf t3 = *(const v8bf*)&Kt[key][32 + kd + 8];
      v16bf b0, b1;
#pragma unroll
      for (int e = 0; e < 8; ++e) {
        b0[e] = t0[e]; b0[e + 8] = t1[e];
        b1[e] = t2[e]; b1[e + 8] = t3[e];
      }
      v8f s = v8f_zero();
      s = wmma_bf16(aq0, b0, s);
      s = wmma_bf16(aq1, b1, s);
#pragma unroll
      for (int r = 0; r < 8; ++r)
        sS[wid][mbase + r][jn * 16 + m16] = s[r] * scale;
    }
    __syncthreads();

    // ---- online softmax (lane = row) ----
    if (lane < 16) {
      float mmax = mrow;
#pragma unroll
      for (int j = 0; j < 32; ++j) mmax = fmaxf(mmax, sS[wid][lane][j]);
      float alpha = __expf(mrow - mmax);
      float ls = 0.0f;
#pragma unroll
      for (int j = 0; j < 32; ++j) {
        float p = __expf(sS[wid][lane][j] - mmax);
        sP[wid][lane][j] = f2bf(p);
        ls += p;
      }
      lrow = lrow * alpha + ls;
      mrow = mmax;
      sAlpha[wid][lane] = alpha;
    }
    __syncthreads();

    // ---- rescale accumulators ----
    float al[8];
#pragma unroll
    for (int r = 0; r < 8; ++r) al[r] = sAlpha[wid][mbase + r];
#pragma unroll
    for (int cg = 0; cg < 4; ++cg)
#pragma unroll
      for (int r = 0; r < 8; ++r) acc[cg][r] *= al[r];

    // ---- P fragment (16x32 bf16) ----
    v16bf ap;
    {
      const int kb = hi * 16;
      v8bf p0 = *(const v8bf*)&sP[wid][m16][kb];
      v8bf p1 = *(const v8bf*)&sP[wid][m16][kb + 8];
#pragma unroll
      for (int e = 0; e < 8; ++e) { ap[e] = p0[e]; ap[e + 8] = p1[e]; }
    }
    // ---- PV: 4 output column groups of 16 (B-frags from LDS V tile) ----
#pragma unroll
    for (int cg = 0; cg < 4; ++cg) {
      v16bf bv;
#pragma unroll
      for (int e = 0; e < 16; ++e) bv[e] = Vt[kd + e][cg * 16 + m16];
      acc[cg] = wmma_bf16(ap, bv, acc[cg]);
    }
    __syncthreads();
  }

  if (lane < 16) sL[wid][lane] = lrow;
  __syncthreads();
  float linv[8];
#pragma unroll
  for (int r = 0; r < 8; ++r) linv[r] = 1.0f / sL[wid][mbase + r];
#pragma unroll
  for (int cg = 0; cg < 4; ++cg)
#pragma unroll
    for (int r = 0; r < 8; ++r)
      o[(long)(r0 + mbase + r) * ldo + cg * 16 + m16] = acc[cg][r] * linv[r];
}

// ---------------------------------------------------------------------------
// small scalar helpers
// ---------------------------------------------------------------------------
__device__ __forceinline__ float block_reduce_sum256(float v, float* red) {
  int tid = threadIdx.x;
  red[tid] = v;
  __syncthreads();
  for (int s = 128; s > 0; s >>= 1) {
    if (tid < s) red[tid] += red[tid + s];
    __syncthreads();
  }
  float r = red[0];
  __syncthreads();
  return r;
}
__device__ __forceinline__ float block_reduce_max256(float v, float* red) {
  int tid = threadIdx.x;
  red[tid] = v;
  __syncthreads();
  for (int s = 128; s > 0; s >>= 1) {
    if (tid < s) red[tid] = fmaxf(red[tid], red[tid + s]);
    __syncthreads();
  }
  float r = red[0];
  __syncthreads();
  return r;
}

// q_sum = sum_query @ wq^T + bq  (1024 outputs; grid 4 x 256)
__global__ __launch_bounds__(256) void qproj_kernel(
    const float* __restrict__ q, const float* __restrict__ w,
    const float* __restrict__ b, float* __restrict__ out) {
  int n = blockIdx.x * 256 + threadIdx.x;
  float v = b[n];
  const float* wr = w + (long)n * 1024;
  for (int k = 0; k < 1024; ++k) v += q[k] * wr[k];
  out[n] = v;
}

// single-query attention per (head, worker); kvs = [4][1024][2048] (K|V)
__global__ __launch_bounds__(256) void sum_attn_kernel(
    const float* __restrict__ qs, const float* __restrict__ kvs,
    float* __restrict__ probs /*[4][16][1024]*/,
    float* __restrict__ ctx /*[4][1024]*/) {
  const int h = blockIdx.x, w = blockIdx.y, tid = threadIdx.x;
  __shared__ float red[256];
  __shared__ float sp[1024];
  const float* q  = qs + h * 64;
  const float* kb = kvs + (long)w * 1024 * 2048;

  float sc[4], lmax = -3.0e38f;
#pragma unroll
  for (int t = 0; t < 4; ++t) {
    int j = tid + t * 256;
    const float* kr = kb + (long)j * 2048 + h * 64;
    float s = 0.0f;
    for (int d = 0; d < 64; ++d) s += q[d] * kr[d];
    sc[t] = s * 0.125f;
    lmax = fmaxf(lmax, sc[t]);
  }
  float bmax = block_reduce_max256(lmax, red);
  float lsum = 0.0f;
#pragma unroll
  for (int t = 0; t < 4; ++t) {
    int j = tid + t * 256;
    float p = __expf(sc[t] - bmax);
    sp[j] = p;
    lsum += p;
  }
  float bsum = block_reduce_sum256(lsum, red);
  float inv = 1.0f / bsum;
#pragma unroll
  for (int t = 0; t < 4; ++t) {
    int j = tid + t * 256;
    float p = sp[j] * inv;
    sp[j] = p;
    probs[((long)w * 16 + h) * 1024 + j] = p;
  }
  __syncthreads();
  for (int d = 0; d < 64; ++d) {
    float part = 0.0f;
#pragma unroll
    for (int t = 0; t < 4; ++t) {
      int j = tid + t * 256;
      part += sp[j] * kb[(long)j * 2048 + 1024 + h * 64 + d];
    }
    float tot = block_reduce_sum256(part, red);
    if (tid == 0) ctx[(long)w * 1024 + h * 64 + d] = tot;
  }
}

// head-averaged weights -> [entropy, gini, max, top5]; bitonic sort in LDS
__global__ __launch_bounds__(256) void stats_kernel(
    const float* __restrict__ probs, float* __restrict__ stats /*[4][4]*/) {
  const int w = blockIdx.x, tid = threadIdx.x;
  __shared__ float arr[1024];
  __shared__ float red[256];
#pragma unroll
  for (int t = 0; t < 4; ++t) {
    int j = tid + t * 256;
    float s = 0.0f;
    for (int h = 0; h < 16; ++h) s += probs[((long)w * 16 + h) * 1024 + j];
    arr[j] = s * (1.0f / 16.0f);
  }
  __syncthreads();
  float e = 0.0f, sm = 0.0f;
#pragma unroll
  for (int t = 0; t < 4; ++t) {
    int j = tid + t * 256;
    float a = arr[j];
    e += a * __logf(a + 1e-8f);
    sm += a;
  }
  float ent  = -block_reduce_sum256(e, red);
  float ssum = block_reduce_sum256(sm, red);
  // bitonic ascending sort of arr[1024]
  for (int k2 = 2; k2 <= 1024; k2 <<= 1) {
    for (int j2 = k2 >> 1; j2 > 0; j2 >>= 1) {
      __syncthreads();
#pragma unroll
      for (int t = 0; t < 4; ++t) {
        int i = tid + t * 256;
        int ixj = i ^ j2;
        if (ixj > i) {
          bool up = ((i & k2) == 0);
          float a = arr[i], b = arr[ixj];
          if ((a > b) == up) { arr[i] = b; arr[ixj] = a; }
        }
      }
    }
  }
  __syncthreads();
  float gn = 0.0f;
#pragma unroll
  for (int t = 0; t < 4; ++t) {
    int j = tid + t * 256;
    gn += (float)(j + 1) * arr[j];
  }
  float gnt = block_reduce_sum256(gn, red);
  if (tid == 0) {
    float gini = 2.0f * gnt / (1024.0f * ssum + 1e-8f) - 1025.0f / 1024.0f;
    float mx = arr[1023];
    float t5 = arr[1023] + arr[1022] + arr[1021] + arr[1020] + arr[1019];
    stats[w * 4 + 0] = ent;
    stats[w * 4 + 1] = gini;
    stats[w * 4 + 2] = mx;
    stats[w * 4 + 3] = t5;
  }
}

// pat MLP + concat + comb projection -> summaries [4][64]
__global__ __launch_bounds__(64) void summary_tail_kernel(
    const float* __restrict__ content, const float* __restrict__ stats,
    const float* __restrict__ pw1, const float* __restrict__ pb1,
    const float* __restrict__ pw2, const float* __restrict__ pb2,
    const float* __restrict__ cw, const float* __restrict__ cb,
    float* __restrict__ summ) {
  const int w = blockIdx.x, n = threadIdx.x;
  __shared__ float h1[64], h2[64];
  float hv = pb1[n];
  for (int k = 0; k < 4; ++k) hv += stats[w * 4 + k] * pw1[n * 4 + k];
  h1[n] = fmaxf(hv, 0.0f);
  __syncthreads();
  float pv = pb2[n];
  for (int k = 0; k < 64; ++k) pv += h1[k] * pw2[n * 64 + k];
  h2[n] = pv;
  __syncthreads();
  float sv = cb[n];
  const float* cwr = cw + (long)n * 1088;
  const float* cont = content + (long)w * 1024;
  for (int k = 0; k < 1024; ++k) sv += cont[k] * cwr[k];
  for (int k = 0; k < 64; ++k) sv += h2[k] * cwr[1024 + k];
  summ[w * 64 + n] = sv;
}

// meta-MHA (4 tokens x 64, 4 heads) + routing MLP + masked renorm -> rm[4][4]
__global__ __launch_bounds__(64) void meta_route_kernel(
    const float* __restrict__ summ, const float* __restrict__ miw,
    const float* __restrict__ mib, const float* __restrict__ mow,
    const float* __restrict__ mob, const float* __restrict__ rw1,
    const float* __restrict__ rb1, const float* __restrict__ rw2,
    const float* __restrict__ rb2, float* __restrict__ rm) {
  __shared__ float qkv[4][192];
  __shared__ float sc[4][4][4];
  __shared__ float oh[4][64];
  __shared__ float ref[4][64];
  __shared__ float hdn[4][64];
  __shared__ float lg[4];
  const int tid = threadIdx.x;
  for (int idx = tid; idx < 768; idx += 64) {
    int t = idx / 192, j = idx % 192;
    float v = mib[j];
    for (int k = 0; k < 64; ++k) v += summ[t * 64 + k] * miw[j * 64 + k];
    qkv[t][j] = v;
  }
  __syncthreads();
  {
    int h = tid >> 4, i = (tid >> 2) & 3, j = tid & 3;
    float s = 0.0f;
    for (int d = 0; d < 16; ++d) s += qkv[i][h * 16 + d] * qkv[j][64 + h * 16 + d];
    sc[h][i][j] = s * 0.25f;  // 1/sqrt(16)
  }
  __syncthreads();
  if (tid < 16) {
    int h = tid >> 2, i = tid & 3;
    float mx = fmaxf(fmaxf(sc[h][i][0], sc[h][i][1]), fmaxf(sc[h][i][2], sc[h][i][3]));
    float s = 0.0f;
    for (int j = 0; j < 4; ++j) { sc[h][i][j] = __expf(sc[h][i][j] - mx); s += sc[h][i][j]; }
    for (int j = 0; j < 4; ++j) sc[h][i][j] /= s;
  }
  __syncthreads();
  for (int idx = tid; idx < 256; idx += 64) {
    int i = idx >> 6, col = idx & 63, h = col >> 4;
    float v = 0.0f;
    for (int j = 0; j < 4; ++j) v += sc[h][i][j] * qkv[j][128 + col];
    oh[i][col] = v;
  }
  __syncthreads();
  for (int idx = tid; idx < 256; idx += 64) {
    int i = idx >> 6, n = idx & 63;
    float v = mob[n];
    for (int k = 0; k < 64; ++k) v += oh[i][k] * mow[n * 64 + k];
    ref[i][n] = v;
  }
  __syncthreads();
  for (int idx = tid; idx < 256; idx += 64) {
    int i = idx >> 6, n = idx & 63;
    float v = rb1[n];
    for (int k = 0; k < 64; ++k) v += ref[i][k] * rw1[n * 64 + k];
    hdn[i][n] = fmaxf(v, 0.0f);
  }
  __syncthreads();
  if (tid < 4) {
    float v = rb2[0];
    for (int k = 0; k < 64; ++k) v += hdn[tid][k] * rw2[k];
    lg[tid] = v;
  }
  __syncthreads();
  if (tid == 0) {
    float mx = fmaxf(fmaxf(lg[0], lg[1]), fmaxf(lg[2], lg[3]));
    float p[4], s = 0.0f;
    for (int i = 0; i < 4; ++i) { p[i] = __expf(lg[i] - mx); s += p[i]; }
    for (int i = 0; i < 4; ++i) p[i] /= s;
    for (int r = 0; r < 4; ++r) {
      float q[4], ss = 0.0f;
      for (int j = 0; j < 4; ++j) { q[j] = (j == r) ? 0.0f : p[j]; ss += q[j]; }
      float inv = 1.0f / (ss + 1e-8f);
      for (int j = 0; j < 4; ++j) rm[r * 4 + j] = q[j] * inv;
    }
  }
}

// kv[r] = sum_w rm[r][w] * local[w]   (grid (4096, 4))
__global__ __launch_bounds__(256) void kvmix_kernel(
    const float* __restrict__ rm, const float* __restrict__ local,
    float* __restrict__ out) {
  const int r = blockIdx.y;
  const long idx = (long)blockIdx.x * 256 + threadIdx.x;
  const float a = rm[r * 4 + 0], b = rm[r * 4 + 1], c = rm[r * 4 + 2],
              d = rm[r * 4 + 3];
  const long N1 = 1048576;
  out[(long)r * N1 + idx] = a * local[idx] + b * local[N1 + idx] +
                            c * local[2 * N1 + idx] + d * local[3 * N1 + idx];
}

// comb = g*cross + (1-g)*local
__global__ __launch_bounds__(256) void gate_combine_kernel(
    const float* __restrict__ g, const float* __restrict__ cross,
    const float* __restrict__ local, float* __restrict__ out, long n) {
  long i = (long)blockIdx.x * 256 + threadIdx.x;
  if (i < n) {
    float gv = g[i];
    out[i] = gv * cross[i] + (1.0f - gv) * local[i];
  }
}

// ---------------------------------------------------------------------------
// launcher
// ---------------------------------------------------------------------------
extern "C" void kernel_launch(void* const* d_in, const int* in_sizes, int n_in,
                              void* d_out, int out_size, void* d_ws,
                              size_t ws_size, hipStream_t stream) {
  (void)in_sizes; (void)n_in; (void)out_size; (void)ws_size;
  const float* x           = (const float*)d_in[0];
  const float* local_in_w  = (const float*)d_in[1];
  const float* local_in_b  = (const float*)d_in[2];
  const float* local_out_w = (const float*)d_in[3];
  const float* local_out_b = (const float*)d_in[4];
  const float* sum_query   = (const float*)d_in[5];
  const float* sum_in_w    = (const float*)d_in[6];
  const float* sum_in_b    = (const float*)d_in[7];
  const float* sum_out_w   = (const float*)d_in[8];
  const float* sum_out_b   = (const float*)d_in[9];
  const float* pat_w1      = (const float*)d_in[10];
  const float* pat_b1      = (const float*)d_in[11];
  const float* pat_w2      = (const float*)d_in[12];
  const float* pat_b2      = (const float*)d_in[13];
  const float* comb_w      = (const float*)d_in[14];
  const float* comb_b      = (const float*)d_in[15];
  const float* meta_in_w   = (const float*)d_in[16];
  const float* meta_in_b   = (const float*)d_in[17];
  const float* meta_out_w  = (const float*)d_in[18];
  const float* meta_out_b  = (const float*)d_in[19];
  const float* route_w1    = (const float*)d_in[20];
  const float* route_b1    = (const float*)d_in[21];
  const float* route_w2    = (const float*)d_in[22];
  const float* route_b2    = (const float*)d_in[23];
  const float* cross_in_w  = (const float*)d_in[24];
  const float* cross_in_b  = (const float*)d_in[25];
  const float* cross_out_w = (const float*)d_in[26];
  const float* cross_out_b = (const float*)d_in[27];
  const float* gate_w      = (const float*)d_in[28];
  const float* gate_b      = (const float*)d_in[29];
  const float* proj_w      = (const float*)d_in[30];
  const float* proj_b      = (const float*)d_in[31];
  float* out = (float*)d_out;

  // workspace arena (floats), 256B aligned
  char* ws = (char*)d_ws;
  size_t off = 0;
  auto alloc = [&](size_t nfloats) -> float* {
    float* p = (float*)(ws + off);
    off += nfloats * sizeof(float);
    off = (off + 255) & ~(size_t)255;
    return p;
  };
  float* bufA    = alloc(12582912);  // bf16 qkv (25MB used); later bf16 crossQ+crossKV
  float* ctxbuf  = alloc(4194304);   // attention context (local, then cross)
  float* localb  = alloc(4194304);   // local MHA output (persistent)
  float* kvsbuf  = alloc(8388608);   // sum K|V; later kvmix + gate
  float* crossb  = alloc(4194304);   // cross MHA output
  float* gtmpb   = alloc(4194304);   // gate pass1; later comb
  float* qs      = alloc(1024);
  float* probs   = alloc(65536);     // [4][16][1024]
  float* ctx_s   = alloc(4096);      // [4][1024]
  float* content = alloc(4096);      // [4][1024]
  float* statsb  = alloc(16);        // [4][4]
  float* summ    = alloc(256);       // [4][64]
  float* rmb     = alloc(16);        // [4][4]

  auto gemm = [&](const float* A, int lda, long sA, const float* Wt, int ldw,
                  const float* bias, const float* Cin, float* C, int ldc,
                  long sC, int M, int N, int K, int act, int batch,
                  int outBf16) {
    dim3 g((N + GN - 1) / GN, (M + GM - 1) / GM, batch);
    gemm_bias_act_kernel<<<g, 256, 0, stream>>>(A, Wt, bias, Cin, C, M, N, K,
                                                lda, ldw, ldc, sA, sC, act,
                                                outBf16);
  };

  // 1. local QKV: [4] 1024x3072, bf16 output for WMMA flash attention
  bf16* qkv_bf = (bf16*)bufA;
  gemm(x, 1024, 1024L * 1024, local_in_w, 1024, local_in_b, nullptr,
       (float*)qkv_bf, 3072, 1024L * 3072, 1024, 3072, 1024, 0, 4, 1);
  // 2. local flash attention
  flash_attn_kernel<<<dim3(16, 16, 4), 128, 0, stream>>>(
      qkv_bf, qkv_bf + 1024, qkv_bf + 2048, ctxbuf, 3072, 3072, 3072, 1024,
      1024L * 3072, 1024L * 3072, 1024L * 3072, 1024L * 1024, 1024, 0.125f);
  // 3. local out projection -> localb
  gemm(ctxbuf, 1024, 1024L * 1024, local_out_w, 1024, local_out_b, nullptr,
       localb, 1024, 1024L * 1024, 1024, 1024, 1024, 0, 4, 0);
  // 4. summarize K|V projections (f32, consumed by scalar kernel)
  gemm(localb, 1024, 1024L * 1024, sum_in_w + 1024L * 1024, 1024,
       sum_in_b + 1024, nullptr, kvsbuf, 2048, 1024L * 2048, 1024, 2048, 1024,
       0, 4, 0);
  // 5. q = sum_query @ wq^T + bq
  qproj_kernel<<<4, 256, 0, stream>>>(sum_query, sum_in_w, sum_in_b, qs);
  // 6. single-query attention per (head, worker)
  sum_attn_kernel<<<dim3(16, 4), 256, 0, stream>>>(qs, kvsbuf, probs, ctx_s);
  // 7. content = ctx_s @ sum_out_w^T + b   (M=4)
  gemm(ctx_s, 1024, 0, sum_out_w, 1024, sum_out_b, nullptr, content, 1024, 0, 4,
       1024, 1024, 0, 1, 0);
  // 8. attn stats (sort + entropy/gini/max/top5)
  stats_kernel<<<4, 256, 0, stream>>>(probs, statsb);
  // 9. pat MLP + comb -> summaries
  summary_tail_kernel<<<4, 64, 0, stream>>>(content, statsb, pat_w1, pat_b1,
                                            pat_w2, pat_b2, comb_w, comb_b,
                                            summ);
  // 10. meta attention + routing -> rm[4][4]
  meta_route_kernel<<<1, 64, 0, stream>>>(summ, meta_in_w, meta_in_b,
                                          meta_out_w, meta_out_b, route_w1,
                                          route_b1, route_w2, route_b2, rmb);
  // 11. kv mixing per rank (reuse kvsbuf)
  float* kvmix = kvsbuf;
  float* gbuf  = kvsbuf + 4194304;
  kvmix_kernel<<<dim3(4096, 4), 256, 0, stream>>>(rmb, localb, kvmix);
  // 12-13. cross Q / KV projections (bf16 outputs, reuse bufA)
  bf16* crossQbf  = (bf16*)bufA;
  bf16* crossKVbf = (bf16*)bufA + 4194304;
  gemm(localb, 1024, 1024L * 1024, cross_in_w, 1024, cross_in_b, nullptr,
       (float*)crossQbf, 1024, 1024L * 1024, 1024, 1024, 1024, 0, 4, 1);
  gemm(kvmix, 1024, 1024L * 1024, cross_in_w + 1024L * 1024, 1024,
       cross_in_b + 1024, nullptr, (float*)crossKVbf, 2048, 1024L * 2048, 1024,
       2048, 1024, 0, 4, 1);
  // 14. cross flash attention
  flash_attn_kernel<<<dim3(16, 16, 4), 128, 0, stream>>>(
      crossQbf, crossKVbf, crossKVbf + 1024, ctxbuf, 1024, 2048, 2048, 1024,
      1024L * 1024, 1024L * 2048, 1024L * 2048, 1024L * 1024, 1024, 0.125f);
  // 15. cross out projection
  gemm(ctxbuf, 1024, 1024L * 1024, cross_out_w, 1024, cross_out_b, nullptr,
       crossb, 1024, 1024L * 1024, 1024, 1024, 1024, 0, 4, 0);
  // 16. gate pass1: lo @ W1^T + b  (gate_w[:, :1024], ldw=2048)
  gemm(localb, 1024, 1024L * 1024, gate_w, 2048, gate_b, nullptr, gtmpb, 1024,
       1024L * 1024, 1024, 1024, 1024, 0, 4, 0);
  // 17. gate pass2: sigmoid(cross @ W2^T + pass1)
  gemm(crossb, 1024, 1024L * 1024, gate_w + 1024, 2048, nullptr, gtmpb, gbuf,
       1024, 1024L * 1024, 1024, 1024, 1024, 2, 4, 0);
  // 18. comb = g*cross + (1-g)*local  (reuse gtmpb)
  float* combb = gtmpb;
  gate_combine_kernel<<<16384, 256, 0, stream>>>(gbuf, crossb, localb, combb,
                                                 4L * 1024 * 1024);
  // 19. final projection -> out
  gemm(combb, 1024, 1024L * 1024, proj_w, 1024, proj_b, nullptr, out, 1024,
       1024L * 1024, 1024, 1024, 1024, 0, 4, 0);
}
